// DroplessMoE_23708219474485
// MI455X (gfx1250) — compile-verified, matrix operand order checked by reading, not run on hardware
//
#include <hip/hip_runtime.h>
#include <hip/hip_bf16.h>

// ---------------- problem constants ----------------
#define T_TOK 8192   // B*S
#define D_DIM 1024
#define F_DIM 4096
#define E_NUM 8

#define BM 128
#define BN 128
#define BK 32
#define LDA 40                     // LDS row stride in bf16: 32 data + 8 pad (16B-aligned)
#define MAXROWS (T_TOK + E_NUM*BM) // 9216: per-expert regions padded to BM

typedef __bf16 bf16_t;
typedef bf16_t v16bf __attribute__((ext_vector_type(16)));
typedef float  v8f   __attribute__((ext_vector_type(8)));
typedef unsigned int v4u __attribute__((ext_vector_type(4)));
typedef int v4i __attribute__((ext_vector_type(4)));
typedef int v8i __attribute__((ext_vector_type(8)));

union Frag { uint4 u[2]; v16bf v; };

// ---------------- TDM availability / arity detection ----------------
#if defined(__has_builtin)
# if __has_builtin(__builtin_amdgcn_tensor_load_to_lds)
#  define HAS_TDM 1
# endif
#endif
#ifndef HAS_TDM
# define HAS_TDM 0
#endif

#if defined(__has_include)
# if __has_include(<hip/amd_detail/amd_gfx1250_TDM.h>)
#  define TDM_ARGS6 1
# endif
#endif
#ifndef TDM_ARGS6
# define TDM_ARGS6 0
#endif

__device__ __forceinline__ void wait_tensorcnt0() {
#if __has_builtin(__builtin_amdgcn_s_wait_tensorcnt)
    __builtin_amdgcn_s_wait_tensorcnt(0);
#else
    asm volatile("s_wait_tensorcnt 0x0" ::: "memory");
#endif
}
__device__ __forceinline__ void wait_tensorcnt2() {
#if __has_builtin(__builtin_amdgcn_s_wait_tensorcnt)
    __builtin_amdgcn_s_wait_tensorcnt(2);
#else
    asm volatile("s_wait_tensorcnt 0x2" ::: "memory");
#endif
}

// Issue one TDM descriptor: 128-row x 32-col bf16 tile, global->LDS, with
// 16B LDS padding after each 64B row (pad_interval=16 DW, pad_amount=4 DW)
// so the LDS image lands with row stride LDA*2 = 80B.
__device__ __forceinline__ void tdm_load_tile_bf16(const bf16_t* gsrc, unsigned ldsOff,
                                                   unsigned tensorW, unsigned tensorH,
                                                   unsigned rowStride /*elements*/) {
#if HAS_TDM
    unsigned long long ga = (unsigned long long)(uintptr_t)gsrc;
    v4u g0;
    g0.x = 1u;                                  // count=1 (valid), user mode
    g0.y = ldsOff;                              // lds_addr (bytes)
    g0.z = (unsigned)ga;                        // global_addr[31:0]
    g0.w = (unsigned)((ga >> 32) & 0x01FFFFFFu) | (2u << 30);  // addr[56:32] | type=2
    v8i g1;
    g1[0] = (int)((1u << 16) | (1u << 20) | (3u << 22) | (3u << 25));
             // data_size=2B | pad_enable | pad_interval=16DW | pad_amount=4DW
    g1[1] = (int)((tensorW & 0xFFFFu) << 16);                    // tensor_dim0[15:0]
    g1[2] = (int)((tensorW >> 16) | ((tensorH & 0xFFFFu) << 16));// dim0 hi | dim1 lo
    g1[3] = (int)((tensorH >> 16) | (32u << 16));                // dim1 hi | tile_dim0=32
    g1[4] = 128;                                                 // tile_dim1=128, tile_dim2=0
    g1[5] = (int)rowStride;                                      // tensor_dim0_stride[31:0]
    g1[6] = 0;                                                   // stride hi, dim1_stride lo
    g1[7] = 0;
    v4i gz = {0, 0, 0, 0};
#if TDM_ARGS6
    v8i gz8 = {0, 0, 0, 0, 0, 0, 0, 0};
    __builtin_amdgcn_tensor_load_to_lds(g0, g1, gz, gz, gz8, 0);
#else
    __builtin_amdgcn_tensor_load_to_lds(g0, g1, gz, gz, 0);
#endif
#else
    (void)gsrc; (void)ldsOff; (void)tensorW; (void)tensorH; (void)rowStride;
#endif
}

// ---------------- init ----------------
__global__ void moe_init_kernel(int* perm, int* cnt, float* zacc, float* wsum) {
    int i = blockIdx.x * 256 + threadIdx.x;
    if (i < MAXROWS) perm[i] = -1;
    if (i < E_NUM) cnt[i] = 0;
    if (i == 0) { *zacc = 0.f; *wsum = 0.f; }
}

// ---------------- router: one wave (32 lanes) per token ----------------
__global__ __launch_bounds__(256) void moe_router_kernel(
    const float* __restrict__ x, const float* __restrict__ gw,
    int* __restrict__ sel, float* __restrict__ wgt,
    int* __restrict__ cnt, float* __restrict__ zacc, float* __restrict__ wsum)
{
    int lane = threadIdx.x & 31;
    int wave = threadIdx.x >> 5;
    int t = blockIdx.x * 8 + wave;
    if (t >= T_TOK) return;

    const float* xr = x + (size_t)t * D_DIM;
    float xv[32];
#pragma unroll
    for (int i = 0; i < 32; ++i) xv[i] = xr[i * 32 + lane];

    float logit[E_NUM];
#pragma unroll
    for (int e = 0; e < E_NUM; ++e) {
        const float* g = gw + (size_t)e * D_DIM;
        float p = 0.f;
#pragma unroll
        for (int i = 0; i < 32; ++i) p += xv[i] * g[i * 32 + lane];
#pragma unroll
        for (int off = 16; off > 0; off >>= 1) p += __shfl_xor(p, off, 32);
        logit[e] = p;
    }
    float m = logit[0]; int am = 0;
#pragma unroll
    for (int e = 1; e < E_NUM; ++e) if (logit[e] > m) { m = logit[e]; am = e; }
    float s = 0.f;
#pragma unroll
    for (int e = 0; e < E_NUM; ++e) s += __expf(logit[e] - m);
    float lse = m + __logf(s);

    if (lane == 0) {
        sel[t] = am;
        wgt[t] = 1.0f;               // k=1: topw / sum(topw) == 1
        atomicAdd(&cnt[am], 1);
        atomicAdd(zacc, lse * lse);
        atomicAdd(wsum, 1.0f);
    }
}

// ---------------- per-expert offsets, padded to BM ----------------
__global__ void moe_offsets_kernel(const int* __restrict__ cnt, int* __restrict__ off,
                                   int* __restrict__ cursor) {
    if (threadIdx.x == 0) {
        int acc = 0;
        for (int e = 0; e < E_NUM; ++e) {
            off[e] = acc; cursor[e] = acc;
            acc += (cnt[e] + BM - 1) / BM * BM;
        }
        off[E_NUM] = acc;
    }
}

// ---------------- scatter token ids into permuted order ----------------
__global__ void moe_scatter_kernel(const int* __restrict__ sel, const float* __restrict__ wgt,
                                   int* __restrict__ cursor, int* __restrict__ perm,
                                   float* __restrict__ rowwgt) {
    int t = blockIdx.x * 256 + threadIdx.x;
    if (t >= T_TOK) return;
    int e = sel[t];
    int pos = atomicAdd(&cursor[e], 1);
    perm[pos] = t;
    rowwgt[pos] = wgt[t];
}

// ---------------- gather + fp32->bf16 pack ----------------
__global__ __launch_bounds__(256) void moe_gather_kernel(
    const float* __restrict__ x, const int* __restrict__ perm, bf16_t* __restrict__ Xp)
{
    int row = blockIdx.x;
    int t = perm[row];
    bf16_t* dst = Xp + (size_t)row * D_DIM + threadIdx.x * 4;
    if (t < 0) {
        dst[0] = (bf16_t)0.f; dst[1] = (bf16_t)0.f; dst[2] = (bf16_t)0.f; dst[3] = (bf16_t)0.f;
    } else {
        const float4 v = *(const float4*)(x + (size_t)t * D_DIM + threadIdx.x * 4);
        dst[0] = (bf16_t)v.x; dst[1] = (bf16_t)v.y; dst[2] = (bf16_t)v.z; dst[3] = (bf16_t)v.w;
    }
}

// ---------------- fp32 -> bf16 weight pre-conversion ----------------
__global__ __launch_bounds__(256) void moe_wcvt_kernel(const float* __restrict__ src,
                                                       bf16_t* __restrict__ dst, size_t n) {
    size_t i = ((size_t)blockIdx.x * 256 + threadIdx.x) * 8;
    if (i >= n) return;
    float4 f0 = *(const float4*)(src + i);
    float4 f1 = *(const float4*)(src + i + 4);
    union { bf16_t h[8]; uint4 u; } pk;
    pk.h[0]=(bf16_t)f0.x; pk.h[1]=(bf16_t)f0.y; pk.h[2]=(bf16_t)f0.z; pk.h[3]=(bf16_t)f0.w;
    pk.h[4]=(bf16_t)f1.x; pk.h[5]=(bf16_t)f1.y; pk.h[6]=(bf16_t)f1.z; pk.h[7]=(bf16_t)f1.w;
    *(uint4*)(dst + i) = pk.u;
}

// ---------------- shared GEMM helpers ----------------
__device__ __forceinline__ int expert_of_rowtile(const int* off, int row0) {
    int e = 0;
#pragma unroll
    for (int i = 0; i < E_NUM; ++i) if (row0 >= off[i]) e = i;
    return e;
}

__device__ __forceinline__ Frag lds_frag_16x32(const bf16_t* base, int lane) {
    // 16-bit A/B fragment layout (ISA 7.12.2): lanes 0-15 hold K[0..7],K[16..23];
    // lanes 16-31 hold K[8..15],K[24..31]; row = lane%16.
    int half = lane >> 4, lr = lane & 15;
    const bf16_t* p = base + lr * LDA + half * 8;
    Frag f;
    f.u[0] = *(const uint4*)(p);
    f.u[1] = *(const uint4*)(p + 16);
    return f;
}

__device__ __forceinline__ float gelu_tanh(float v) {
    float u = 0.7978845608f * (v + 0.044715f * v * v * v);
    return 0.5f * v * (1.0f + tanhf(u));
}

// synchronous bf16 tile staging (non-TDM fallback), row stride in elements
__device__ __forceinline__ void stage_tile_bf16(const bf16_t* src, size_t stride,
                                                bf16_t* dst, int tid) {
#pragma unroll
    for (int j = 0; j < 2; ++j) {
        int cc = tid * 2 + j;
        int r = cc >> 2, seg = cc & 3;
        *(uint4*)(dst + r * LDA + seg * 8) = *(const uint4*)(src + (size_t)r * stride + seg * 8);
    }
}

// ============================================================================
// TDM-staged GEMMs (bf16 A and bf16 pre-converted weights, double buffered)
// ============================================================================
__global__ __launch_bounds__(256) void moe_gemm1_tdm(
    const bf16_t* __restrict__ Xp, const bf16_t* __restrict__ w1b,
    const float* __restrict__ b1, bf16_t* __restrict__ H, const int* __restrict__ off)
{
    __shared__ bf16_t sA[2][BM * LDA];
    __shared__ bf16_t sB[2][BN * LDA];

    int row0 = blockIdx.y * BM;
    if (row0 >= off[E_NUM]) return;
    int e = expert_of_rowtile(off, row0);
    int n0 = blockIdx.x * BN;

    const bf16_t* aT = Xp + (size_t)row0 * D_DIM;
    const bf16_t* bT = w1b + (size_t)e * F_DIM * D_DIM + (size_t)n0 * D_DIM;

    int tid = threadIdx.x;
    int lane = tid & 31, wave = tid >> 5;
    int wm = wave & 1, wn = wave >> 1;

    const v8f vzero = {0.f,0.f,0.f,0.f,0.f,0.f,0.f,0.f};
    v8f acc[4][2];
#pragma unroll
    for (int ti = 0; ti < 4; ++ti)
#pragma unroll
        for (int tj = 0; tj < 2; ++tj) acc[ti][tj] = vzero;

#if HAS_TDM
    if (wave == 0) {
        tdm_load_tile_bf16(aT, (unsigned)(uintptr_t)&sA[0][0], D_DIM, MAXROWS - row0, D_DIM);
        tdm_load_tile_bf16(bT, (unsigned)(uintptr_t)&sB[0][0], D_DIM, F_DIM - n0, D_DIM);
    }
#endif
    int cur = 0;
    for (int k0 = 0; k0 < D_DIM; k0 += BK) {
#if HAS_TDM
        if (wave == 0) {
            if (k0 + BK < D_DIM) {
                tdm_load_tile_bf16(aT + k0 + BK, (unsigned)(uintptr_t)&sA[cur ^ 1][0],
                                   D_DIM, MAXROWS - row0, D_DIM);
                tdm_load_tile_bf16(bT + k0 + BK, (unsigned)(uintptr_t)&sB[cur ^ 1][0],
                                   D_DIM, F_DIM - n0, D_DIM);
                wait_tensorcnt2();   // current buffer's 2 DMAs complete
            } else {
                wait_tensorcnt0();
            }
        }
#else
        stage_tile_bf16(aT + k0, D_DIM, &sA[cur][0], tid);
        stage_tile_bf16(bT + k0, D_DIM, &sB[cur][0], tid);
#endif
        __syncthreads();

        Frag a[4], b[2];
#pragma unroll
        for (int ti = 0; ti < 4; ++ti)
            a[ti] = lds_frag_16x32(&sA[cur][0] + (wm * 64 + ti * 16) * LDA, lane);
#pragma unroll
        for (int tj = 0; tj < 2; ++tj)
            b[tj] = lds_frag_16x32(&sB[cur][0] + (wn * 32 + tj * 16) * LDA, lane);
#pragma unroll
        for (int ti = 0; ti < 4; ++ti)
#pragma unroll
            for (int tj = 0; tj < 2; ++tj)
                acc[ti][tj] = __builtin_amdgcn_wmma_f32_16x16x32_bf16(
                    false, a[ti].v, false, b[tj].v, (short)0, acc[ti][tj], false, false);
        __syncthreads();   // all waves done with buf before DMA reuses it
        cur ^= 1;
    }

    const float* bias = b1 + (size_t)e * F_DIM;
    int half = lane >> 4, lc = lane & 15;
#pragma unroll
    for (int ti = 0; ti < 4; ++ti)
#pragma unroll
        for (int tj = 0; tj < 2; ++tj) {
            int n = n0 + wn * 32 + tj * 16 + lc;
            float bv = bias[n];
#pragma unroll
            for (int r = 0; r < 8; ++r) {
                int rowl = wm * 64 + ti * 16 + half * 8 + r;
                H[(size_t)(row0 + rowl) * F_DIM + n] = (bf16_t)gelu_tanh(acc[ti][tj][r] + bv);
            }
        }
}

__global__ __launch_bounds__(256) void moe_gemm2_tdm(
    const bf16_t* __restrict__ H, const bf16_t* __restrict__ w2b,
    const float* __restrict__ b2, float* __restrict__ dout,
    const int* __restrict__ off, const int* __restrict__ perm,
    const float* __restrict__ rowwgt)
{
    __shared__ bf16_t sA[2][BM * LDA];
    __shared__ bf16_t sB[2][BN * LDA];

    int row0 = blockIdx.y * BM;
    if (row0 >= off[E_NUM]) return;
    int e = expert_of_rowtile(off, row0);
    int n0 = blockIdx.x * BN;

    const bf16_t* aT = H + (size_t)row0 * F_DIM;
    const bf16_t* bT = w2b + (size_t)e * D_DIM * F_DIM + (size_t)n0 * F_DIM;

    int tid = threadIdx.x;
    int lane = tid & 31, wave = tid >> 5;
    int wm = wave & 1, wn = wave >> 1;

    const v8f vzero = {0.f,0.f,0.f,0.f,0.f,0.f,0.f,0.f};
    v8f acc[4][2];
#pragma unroll
    for (int ti = 0; ti < 4; ++ti)
#pragma unroll
        for (int tj = 0; tj < 2; ++tj) acc[ti][tj] = vzero;

#if HAS_TDM
    if (wave == 0) {
        tdm_load_tile_bf16(aT, (unsigned)(uintptr_t)&sA[0][0], F_DIM, MAXROWS - row0, F_DIM);
        tdm_load_tile_bf16(bT, (unsigned)(uintptr_t)&sB[0][0], F_DIM, D_DIM - n0, F_DIM);
    }
#endif
    int cur = 0;
    for (int k0 = 0; k0 < F_DIM; k0 += BK) {
#if HAS_TDM
        if (wave == 0) {
            if (k0 + BK < F_DIM) {
                tdm_load_tile_bf16(aT + k0 + BK, (unsigned)(uintptr_t)&sA[cur ^ 1][0],
                                   F_DIM, MAXROWS - row0, F_DIM);
                tdm_load_tile_bf16(bT + k0 + BK, (unsigned)(uintptr_t)&sB[cur ^ 1][0],
                                   F_DIM, D_DIM - n0, F_DIM);
                wait_tensorcnt2();
            } else {
                wait_tensorcnt0();
            }
        }
#else
        stage_tile_bf16(aT + k0, F_DIM, &sA[cur][0], tid);
        stage_tile_bf16(bT + k0, F_DIM, &sB[cur][0], tid);
#endif
        __syncthreads();

        Frag a[4], b[2];
#pragma unroll
        for (int ti = 0; ti < 4; ++ti)
            a[ti] = lds_frag_16x32(&sA[cur][0] + (wm * 64 + ti * 16) * LDA, lane);
#pragma unroll
        for (int tj = 0; tj < 2; ++tj)
            b[tj] = lds_frag_16x32(&sB[cur][0] + (wn * 32 + tj * 16) * LDA, lane);
#pragma unroll
        for (int ti = 0; ti < 4; ++ti)
#pragma unroll
            for (int tj = 0; tj < 2; ++tj)
                acc[ti][tj] = __builtin_amdgcn_wmma_f32_16x16x32_bf16(
                    false, a[ti].v, false, b[tj].v, (short)0, acc[ti][tj], false, false);
        __syncthreads();
        cur ^= 1;
    }

    const float* bias = b2 + (size_t)e * D_DIM;
    int half = lane >> 4, lc = lane & 15;
#pragma unroll
    for (int ti = 0; ti < 4; ++ti)
#pragma unroll
        for (int tj = 0; tj < 2; ++tj) {
            int n = n0 + wn * 32 + tj * 16 + lc;
            float bv = bias[n];
#pragma unroll
            for (int r = 0; r < 8; ++r) {
                int rowl = wm * 64 + ti * 16 + half * 8 + r;
                int rowg = row0 + rowl;
                int t = perm[rowg];
                if (t >= 0)
                    dout[(size_t)t * D_DIM + n] = rowwgt[rowg] * (acc[ti][tj][r] + bv);
            }
        }
}

// ============================================================================
// Fallback GEMMs (fp32 weights converted in-loop) — used if ws is small
// ============================================================================
__global__ __launch_bounds__(256) void moe_gemm1_f32(
    const bf16_t* __restrict__ Xp, const float* __restrict__ w1,
    const float* __restrict__ b1, bf16_t* __restrict__ H, const int* __restrict__ off)
{
    __shared__ bf16_t sA[BM * LDA];
    __shared__ bf16_t sB[BN * LDA];

    int row0 = blockIdx.y * BM;
    if (row0 >= off[E_NUM]) return;
    int e = expert_of_rowtile(off, row0);
    int n0 = blockIdx.x * BN;
    const float* w = w1 + (size_t)e * F_DIM * D_DIM;

    int tid = threadIdx.x;
    int lane = tid & 31, wave = tid >> 5;
    int wm = wave & 1, wn = wave >> 1;

    const v8f vzero = {0.f,0.f,0.f,0.f,0.f,0.f,0.f,0.f};
    v8f acc[4][2];
#pragma unroll
    for (int ti = 0; ti < 4; ++ti)
#pragma unroll
        for (int tj = 0; tj < 2; ++tj) acc[ti][tj] = vzero;

    for (int k0 = 0; k0 < D_DIM; k0 += BK) {
        stage_tile_bf16(Xp + (size_t)row0 * D_DIM + k0, D_DIM, sA, tid);
#pragma unroll
        for (int j = 0; j < 2; ++j) {
            int cc = tid * 2 + j;
            int r = cc >> 2, seg = cc & 3;
            const float4* sp = (const float4*)(w + (size_t)(n0 + r) * D_DIM + k0 + seg * 8);
            float4 f0 = sp[0], f1 = sp[1];
            bf16_t* d = sB + r * LDA + seg * 8;
            d[0]=(bf16_t)f0.x; d[1]=(bf16_t)f0.y; d[2]=(bf16_t)f0.z; d[3]=(bf16_t)f0.w;
            d[4]=(bf16_t)f1.x; d[5]=(bf16_t)f1.y; d[6]=(bf16_t)f1.z; d[7]=(bf16_t)f1.w;
        }
        if (k0 + BK < D_DIM)
            __builtin_prefetch(w + (size_t)(n0 + (tid >> 1)) * D_DIM + k0 + BK, 0, 3);
        __syncthreads();

        Frag a[4], b[2];
#pragma unroll
        for (int ti = 0; ti < 4; ++ti)
            a[ti] = lds_frag_16x32(sA + (wm * 64 + ti * 16) * LDA, lane);
#pragma unroll
        for (int tj = 0; tj < 2; ++tj)
            b[tj] = lds_frag_16x32(sB + (wn * 32 + tj * 16) * LDA, lane);
#pragma unroll
        for (int ti = 0; ti < 4; ++ti)
#pragma unroll
            for (int tj = 0; tj < 2; ++tj)
                acc[ti][tj] = __builtin_amdgcn_wmma_f32_16x16x32_bf16(
                    false, a[ti].v, false, b[tj].v, (short)0, acc[ti][tj], false, false);
        __syncthreads();
    }

    const float* bias = b1 + (size_t)e * F_DIM;
    int half = lane >> 4, lc = lane & 15;
#pragma unroll
    for (int ti = 0; ti < 4; ++ti)
#pragma unroll
        for (int tj = 0; tj < 2; ++tj) {
            int n = n0 + wn * 32 + tj * 16 + lc;
            float bv = bias[n];
#pragma unroll
            for (int r = 0; r < 8; ++r) {
                int rowl = wm * 64 + ti * 16 + half * 8 + r;
                H[(size_t)(row0 + rowl) * F_DIM + n] = (bf16_t)gelu_tanh(acc[ti][tj][r] + bv);
            }
        }
}

__global__ __launch_bounds__(256) void moe_gemm2_f32(
    const bf16_t* __restrict__ H, const float* __restrict__ w2,
    const float* __restrict__ b2, float* __restrict__ dout,
    const int* __restrict__ off, const int* __restrict__ perm,
    const float* __restrict__ rowwgt)
{
    __shared__ bf16_t sA[BM * LDA];
    __shared__ bf16_t sB[BN * LDA];

    int row0 = blockIdx.y * BM;
    if (row0 >= off[E_NUM]) return;
    int e = expert_of_rowtile(off, row0);
    int n0 = blockIdx.x * BN;
    const float* w = w2 + (size_t)e * D_DIM * F_DIM;

    int tid = threadIdx.x;
    int lane = tid & 31, wave = tid >> 5;
    int wm = wave & 1, wn = wave >> 1;

    const v8f vzero = {0.f,0.f,0.f,0.f,0.f,0.f,0.f,0.f};
    v8f acc[4][2];
#pragma unroll
    for (int ti = 0; ti < 4; ++ti)
#pragma unroll
        for (int tj = 0; tj < 2; ++tj) acc[ti][tj] = vzero;

    for (int k0 = 0; k0 < F_DIM; k0 += BK) {
        stage_tile_bf16(H + (size_t)row0 * F_DIM + k0, F_DIM, sA, tid);
#pragma unroll
        for (int j = 0; j < 2; ++j) {
            int cc = tid * 2 + j;
            int r = cc >> 2, seg = cc & 3;
            const float4* sp = (const float4*)(w + (size_t)(n0 + r) * F_DIM + k0 + seg * 8);
            float4 f0 = sp[0], f1 = sp[1];
            bf16_t* d = sB + r * LDA + seg * 8;
            d[0]=(bf16_t)f0.x; d[1]=(bf16_t)f0.y; d[2]=(bf16_t)f0.z; d[3]=(bf16_t)f0.w;
            d[4]=(bf16_t)f1.x; d[5]=(bf16_t)f1.y; d[6]=(bf16_t)f1.z; d[7]=(bf16_t)f1.w;
        }
        if (k0 + BK < F_DIM)
            __builtin_prefetch(w + (size_t)(n0 + (tid >> 1)) * F_DIM + k0 + BK, 0, 3);
        __syncthreads();

        Frag a[4], b[2];
#pragma unroll
        for (int ti = 0; ti < 4; ++ti)
            a[ti] = lds_frag_16x32(sA + (wm * 64 + ti * 16) * LDA, lane);
#pragma unroll
        for (int tj = 0; tj < 2; ++tj)
            b[tj] = lds_frag_16x32(sB + (wn * 32 + tj * 16) * LDA, lane);
#pragma unroll
        for (int ti = 0; ti < 4; ++ti)
#pragma unroll
            for (int tj = 0; tj < 2; ++tj)
                acc[ti][tj] = __builtin_amdgcn_wmma_f32_16x16x32_bf16(
                    false, a[ti].v, false, b[tj].v, (short)0, acc[ti][tj], false, false);
        __syncthreads();
    }

    const float* bias = b2 + (size_t)e * D_DIM;
    int half = lane >> 4, lc = lane & 15;
#pragma unroll
    for (int ti = 0; ti < 4; ++ti)
#pragma unroll
        for (int tj = 0; tj < 2; ++tj) {
            int n = n0 + wn * 32 + tj * 16 + lc;
            float bv = bias[n];
#pragma unroll
            for (int r = 0; r < 8; ++r) {
                int rowl = wm * 64 + ti * 16 + half * 8 + r;
                int rowg = row0 + rowl;
                int t = perm[rowg];
                if (t >= 0)
                    dout[(size_t)t * D_DIM + n] = rowwgt[rowg] * (acc[ti][tj][r] + bv);
            }
        }
}

// ---------------- losses ----------------
__global__ void moe_finalize_kernel(const float* __restrict__ zacc,
                                    const float* __restrict__ wsum,
                                    float* __restrict__ dout) {
    if (threadIdx.x == 0) {
        float rppe = wsum[0] / (float)T_TOK;           // mean(topw) (==1.0 for k=1)
        dout[(size_t)T_TOK * D_DIM]     = (float)E_NUM * rppe;    // aux loss
        dout[(size_t)T_TOK * D_DIM + 1] = zacc[0] / (float)T_TOK; // z loss
    }
}

// ---------------- launcher ----------------
extern "C" void kernel_launch(void* const* d_in, const int* in_sizes, int n_in,
                              void* d_out, int out_size, void* d_ws, size_t ws_size,
                              hipStream_t stream) {
    const float* x  = (const float*)d_in[0];
    const float* gw = (const float*)d_in[1];
    const float* w1 = (const float*)d_in[2];
    const float* b1 = (const float*)d_in[3];
    const float* w2 = (const float*)d_in[4];
    const float* b2 = (const float*)d_in[5];
    float* out = (float*)d_out;

    char* ws = (char*)d_ws;
    int*   cnt    = (int*)ws;              // 8
    int*   off    = cnt + 8;               // 9
    int*   cursor = off + 16;              // 8
    float* zacc   = (float*)(cursor + 8);
    float* wsum   = zacc + 1;
    int*   sel    = (int*)(ws + 256);
    float* wgt    = (float*)(ws + 256 + 4 * (size_t)T_TOK);
    int*   perm   = (int*)(ws + 256 + 8 * (size_t)T_TOK);
    float* rowwgt = (float*)(ws + 256 + 8 * (size_t)T_TOK + 4 * (size_t)MAXROWS);
    size_t big0   = (256 + 8 * (size_t)T_TOK + 8 * (size_t)MAXROWS + 255) & ~(size_t)255;
    bf16_t* Xp    = (bf16_t*)(ws + big0);
    bf16_t* H     = Xp + (size_t)MAXROWS * D_DIM;

    size_t wElems   = (size_t)E_NUM * F_DIM * D_DIM;          // per weight tensor
    size_t baseNeed = big0 + ((size_t)MAXROWS * D_DIM + (size_t)MAXROWS * F_DIM) * 2;
    size_t fullNeed = baseNeed + 2 * wElems * 2;              // + bf16 w1 + bf16 w2
    bf16_t* w1b = (bf16_t*)(ws + baseNeed);
    bf16_t* w2b = w1b + wElems;
    bool useBf16W = (ws_size >= fullNeed);                    // fixed per run -> deterministic

    moe_init_kernel<<<(MAXROWS + 255) / 256, 256, 0, stream>>>(perm, cnt, zacc, wsum);
    moe_router_kernel<<<T_TOK / 8, 256, 0, stream>>>(x, gw, sel, wgt, cnt, zacc, wsum);
    moe_offsets_kernel<<<1, 32, 0, stream>>>(cnt, off, cursor);
    moe_scatter_kernel<<<T_TOK / 256, 256, 0, stream>>>(sel, wgt, cursor, perm, rowwgt);
    moe_gather_kernel<<<MAXROWS, 256, 0, stream>>>(x, perm, Xp);

    dim3 g1(F_DIM / BN, MAXROWS / BM);  // 32 x 72
    dim3 g2(D_DIM / BN, MAXROWS / BM);  // 8 x 72
    if (useBf16W) {
        int cvtBlocks = (int)(wElems / (8 * 256));
        moe_wcvt_kernel<<<cvtBlocks, 256, 0, stream>>>(w1, w1b, wElems);
        moe_wcvt_kernel<<<cvtBlocks, 256, 0, stream>>>(w2, w2b, wElems);
        moe_gemm1_tdm<<<g1, 256, 0, stream>>>(Xp, w1b, b1, H, off);
        moe_gemm2_tdm<<<g2, 256, 0, stream>>>(H, w2b, b2, out, off, perm, rowwgt);
    } else {
        moe_gemm1_f32<<<g1, 256, 0, stream>>>(Xp, w1, b1, H, off);
        moe_gemm2_f32<<<g2, 256, 0, stream>>>(H, w2, b2, out, off, perm, rowwgt);
    }
    moe_finalize_kernel<<<1, 32, 0, stream>>>(zacc, wsum, out);
}
